// graph_one_ring_conv_26388279067292
// MI455X (gfx1250) — matrix operand order repaired and checked by reading, not compiled, not run on hardware
//
#include <hip/hip_runtime.h>

// One-ring graph conv: y = gather(x, neigh).reshape(N, 896) @ W^T + b
// bf16 WMMA; x converted to bf16 once (42MB, L2-resident); W pre-swizzled and
// streamed through double-buffered LDS with GLOBAL_LOAD_ASYNC_TO_LDS_B128
// (ASYNCcnt) so staging never interferes with the A-gather LOADcnt pipeline.

#define IN_F   128
#define OUT_F  128
#define KNBR   7
#define KTOT   (KNBR * IN_F)   // 896
#define NCT    (OUT_F / 16)    // 8 column tiles
// Wp: per-j block of 8192 u32 (32KB); within a block:
//   u32 idx = half*4096 + ((tt*8 + c)*32 + lane)*4 + q,  VGPR r = half*4 + q
#define WP_U32_PER_J   8192
#define WP_U32_TOTAL   (KNBR * WP_U32_PER_J)   // 57344
#define LDS_CHUNKS_PER_BUF 2048                 // uint4 chunks per 32KB buffer

#define HAS_ASYNC_LDS __has_builtin(__builtin_amdgcn_global_load_async_to_lds_b128)

typedef __bf16 bf16_t;
typedef bf16_t v16bf __attribute__((ext_vector_type(16)));
typedef float  v8f   __attribute__((ext_vector_type(8)));
typedef int    v4i   __attribute__((ext_vector_type(4)));
typedef __attribute__((address_space(1))) v4i gv4i;   // global-AS 16B chunk
typedef __attribute__((address_space(3))) v4i lv4i;   // LDS-AS   16B chunk

union FragAB { uint4 u[2]; v16bf v; };

__device__ __forceinline__ gv4i* to_global_as(const void* p) {
    return (gv4i*)(unsigned long long)(uintptr_t)p;
}
__device__ __forceinline__ lv4i* to_lds_as(const void* p) {
    // Flat LDS address: low 32 bits are the LDS byte offset (ISA 10.2).
    return (lv4i*)(unsigned)(uintptr_t)p;
}

__device__ __forceinline__ unsigned pack_bf16(float a, float b) {
    unsigned ua = __builtin_bit_cast(unsigned, a);
    unsigned ub = __builtin_bit_cast(unsigned, b);
    unsigned ra = (ua + 0x7FFFu + ((ua >> 16) & 1u)) >> 16;   // RNE
    unsigned rb = (ub + 0x7FFFu + ((ub >> 16) & 1u)) >> 16;
    return (ra & 0xFFFFu) | (rb << 16);
}

__device__ __forceinline__ void wait_asynccnt0() {
#if __has_builtin(__builtin_amdgcn_s_wait_asynccnt)
    __builtin_amdgcn_s_wait_asynccnt(0);
#else
    asm volatile("s_wait_asynccnt 0x0" ::: "memory");
#endif
}

// ---- Kernel 1: x (f32) -> x_bf16 ------------------------------------------
__global__ void cvt_x_kernel(const float* __restrict__ x,
                             uint2* __restrict__ xb, int nvec4) {
    int tid = blockIdx.x * blockDim.x + threadIdx.x;
    if (tid >= nvec4) return;
    float4 f = ((const float4*)x)[tid];
    uint2 o;
    o.x = pack_bf16(f.x, f.y);
    o.y = pack_bf16(f.z, f.w);
    xb[tid] = o;
}

// ---- Kernel 2: W [128, 896] f32 -> pre-swizzled bf16 B fragments ----------
// Per-lane B fragment (16x16x32 bf16): lanes 0..15 hold K=0..15 of column
// n=c*16+lane (VGPR r = K pair 2r,2r+1); lanes 16..31 hold K=16..31.
// Fragment halves live in separate 16KB halves so ds_load_b128 has 16B lane
// stride (conflict-minimal), and each 32KB j-block is a linear LDS copy.
__global__ void pack_w_kernel(const float* __restrict__ W,
                              unsigned* __restrict__ Wp) {
    int tid = blockIdx.x * blockDim.x + threadIdx.x;   // < 57344
    int j    = tid >> 13;
    int r13  = tid & 8191;
    int half = r13 >> 12;
    int r12  = r13 & 4095;
    int f4   = r12 >> 2;           // (tt*8 + c)*32 + lane
    int q    = r12 & 3;
    int tt   = f4 >> 8;
    int c    = (f4 >> 5) & 7;
    int lane = f4 & 31;
    int r    = half * 4 + q;       // fragment VGPR index 0..7
    int n    = c * 16 + (lane & 15);
    int k    = (j * 4 + tt) * 32 + ((lane >> 4) << 4) + 2 * r;
    Wp[tid] = pack_bf16(W[n * KTOT + k], W[n * KTOT + k + 1]);
}

// ---- Kernel 3: gathered GEMM ----------------------------------------------
// Block = 8 waves, each wave a 16-row x 128-col strip (block covers 128 rows).
// B streamed global->LDS double-buffered per neighbor j; all waves share it.
__global__ void __launch_bounds__(256)
ring_gemm_kernel(const uint4* __restrict__ xb4,     // x in bf16, 16B chunks
                 const int*  __restrict__ neigh,    // [N, 7]
                 const uint4* __restrict__ wp4,     // pre-swizzled W (uint4)
                 const float* __restrict__ bias,    // [128]
                 float* __restrict__ out,           // [N, 128]
                 int N) {
    extern __shared__ uint4 lds_buf[];              // 2 x 2048 uint4 = 64KB

    const int tid  = threadIdx.x;
    const int lane = tid & 31;
    const int wave = tid >> 5;
    const int tile = blockIdx.x * 8 + wave;

    const int m0 = tile * 16;
    const int hs = lane >> 4;        // which K-half this lane holds (A layout)
    const int lm = lane & 15;        // A row within tile / output column
    int row = m0 + lm;
    if (row >= N) row = N - 1;       // clamp (tail waves compute, stores masked)

    // Neighbor row bases (uint4-chunk index into x_bf16), all 7 up front.
    int rc[KNBR];
#pragma unroll
    for (int j = 0; j < KNBR; ++j)
        rc[j] = neigh[row * KNBR + j] * (IN_F * 2 / 16);

    // A-fragment loader (j, tt static after unrolling).
    auto loadA = [&](int j, int tt) {
        FragAB a;
        a.u[0] = xb4[rc[j] + tt * 4 + 2 * hs];      // K 0..7   (or 16..23)
        a.u[1] = xb4[rc[j] + tt * 4 + 2 * hs + 1];  // K 8..15  (or 24..31)
        return a;
    };

    // Accumulators = bias broadcast (C value depends only on column).
    v8f acc[NCT];
#pragma unroll
    for (int c = 0; c < NCT; ++c) {
        float bv = bias[c * 16 + lm];
#pragma unroll
        for (int r = 0; r < 8; ++r) acc[c][r] = bv;
    }

    // Prefetch first A fragment.
    FragAB a_cur = loadA(0, 0);

    // Prime LDS buffer 0 with B(j=0).
#if HAS_ASYNC_LDS
    // Same 24-bit immediate applies to both LDS and global sides; bases carry
    // the per-thread and per-buffer components, the imm carries s*4096.
#define STAGE_ASYNC(gp, lp)                                                    \
    do {                                                                       \
        __builtin_amdgcn_global_load_async_to_lds_b128((gp), (lp), 0*4096, 0); \
        __builtin_amdgcn_global_load_async_to_lds_b128((gp), (lp), 1*4096, 0); \
        __builtin_amdgcn_global_load_async_to_lds_b128((gp), (lp), 2*4096, 0); \
        __builtin_amdgcn_global_load_async_to_lds_b128((gp), (lp), 3*4096, 0); \
        __builtin_amdgcn_global_load_async_to_lds_b128((gp), (lp), 4*4096, 0); \
        __builtin_amdgcn_global_load_async_to_lds_b128((gp), (lp), 5*4096, 0); \
        __builtin_amdgcn_global_load_async_to_lds_b128((gp), (lp), 6*4096, 0); \
        __builtin_amdgcn_global_load_async_to_lds_b128((gp), (lp), 7*4096, 0); \
    } while (0)
    {
        gv4i* gp0 = to_global_as(wp4 + tid);
        lv4i* lp0 = to_lds_as(lds_buf + tid);
        STAGE_ASYNC(gp0, lp0);
    }
    wait_asynccnt0();
#else
#pragma unroll
    for (int s = 0; s < 8; ++s)
        lds_buf[tid + s * 256] = wp4[tid + s * 256];
#endif
    __syncthreads();

#pragma unroll
    for (int j = 0; j < KNBR; ++j) {
#if HAS_ASYNC_LDS
        // Kick off async staging of B(j+1) straight into the other LDS buffer.
        // ASYNCcnt-tracked: zero VGPR cost, no LOADcnt interference with the
        // A-gather prefetch stream.
        if (j < KNBR - 1) {
            gv4i* gp = to_global_as(wp4 + (j + 1) * LDS_CHUNKS_PER_BUF + tid);
            lv4i* lp = to_lds_as(lds_buf + ((j + 1) & 1) * LDS_CHUNKS_PER_BUF + tid);
            STAGE_ASYNC(gp, lp);
        }
#else
        uint4 stage[8];
        if (j < KNBR - 1) {
#pragma unroll
            for (int s = 0; s < 8; ++s)
                stage[s] = wp4[(j + 1) * LDS_CHUNKS_PER_BUF + tid + s * 256];
        }
#endif

        const uint4* wbuf = lds_buf + (j & 1) * LDS_CHUNKS_PER_BUF;
#pragma unroll
        for (int tt = 0; tt < 4; ++tt) {
            // Prefetch A for the next 8-WMMA group (double buffer).
            FragAB a_nxt;
            if (!(j == KNBR - 1 && tt == 3)) {
                const int nj  = (tt == 3) ? j + 1 : j;
                const int ntt = (tt == 3) ? 0 : tt + 1;
                a_nxt = loadA(nj, ntt);
            }
#pragma unroll
            for (int c = 0; c < NCT; ++c) {
                FragAB b;                           // B fragment from LDS
                const int f4i = (tt * 8 + c) * 32 + lane;
                b.u[0] = wbuf[f4i];
                b.u[1] = wbuf[1024 + f4i];
                acc[c] = __builtin_amdgcn_wmma_f32_16x16x32_bf16(
                    false, a_cur.v, false, b.v, (short)0, acc[c], false, false);
            }
            a_cur = a_nxt;
        }

        // Publish B(j+1); barrier also guarantees everyone finished reading
        // buffer (j&1) before it is overwritten in iteration j+1.
        if (j < KNBR - 1) {
#if HAS_ASYNC_LDS
            wait_asynccnt0();
#else
            uint4* nbuf = lds_buf + ((j + 1) & 1) * LDS_CHUNKS_PER_BUF;
#pragma unroll
            for (int s = 0; s < 8; ++s)
                nbuf[tid + s * 256] = stage[s];
#endif
            __syncthreads();
        }
    }

    // Store: D layout -> row = m0 + r + 8*hs, col = c*16 + lm (NT stores).
#pragma unroll
    for (int c = 0; c < NCT; ++c) {
        const int col = c * 16 + lm;
#pragma unroll
        for (int r = 0; r < 8; ++r) {
            const int orow = m0 + hs * 8 + r;
            if (orow < N)
                __builtin_nontemporal_store(acc[c][r], &out[orow * OUT_F + col]);
        }
    }
}

extern "C" void kernel_launch(void* const* d_in, const int* in_sizes, int n_in,
                              void* d_out, int out_size, void* d_ws, size_t ws_size,
                              hipStream_t stream) {
    const float* x     = (const float*)d_in[0];
    const int*   neigh = (const int*)d_in[1];
    const float* W     = (const float*)d_in[2];
    const float* bias  = (const float*)d_in[3];
    float*       out   = (float*)d_out;

    const int N = in_sizes[0] / IN_F;                 // 163842

    // Workspace: [ x_bf16 : N*128*2 bytes ][ Wp : 57344*4 = 229376 bytes ]
    unsigned* xb = (unsigned*)d_ws;
    unsigned* Wp = (unsigned*)((char*)d_ws + (size_t)N * IN_F * 2);

    // 1) convert x -> bf16
    {
        int nvec4 = (N * IN_F) / 4;
        int blocks = (nvec4 + 255) / 256;
        cvt_x_kernel<<<blocks, 256, 0, stream>>>(x, (uint2*)xb, nvec4);
    }
    // 2) pre-swizzle W into LDS-streamable B-fragment layout
    {
        pack_w_kernel<<<WP_U32_TOTAL / 256, 256, 0, stream>>>(W, Wp);
    }
    // 3) gathered GEMM with bf16 WMMA, B via double-buffered (async) LDS
    {
        int numTiles = (N + 15) / 16;                 // 10241
        int blocks = (numTiles + 7) / 8;              // 8 waves / block
        ring_gemm_kernel<<<blocks, 256, 65536, stream>>>(
            (const uint4*)xb, neigh, (const uint4*)Wp, bias, out, N);
    }
}
